// SuperDocEncoder_30322469109908
// MI455X (gfx1250) — matrix operand order repaired and checked by reading, not compiled, not run on hardware
//
#include <hip/hip_runtime.h>

typedef __bf16 bf16;
typedef __attribute__((ext_vector_type(16))) __bf16 v16bf;
typedef __attribute__((ext_vector_type(8)))  __bf16 v8bf;
typedef __attribute__((ext_vector_type(8)))  float  v8f;

union FragBF { v16bf v; v8bf h[2]; };

#define KB  32   // K elements consumed per WMMA step (bf16)
#define KBP 40   // padded LDS row stride in elements (80B -> conflict-free b128 gathers)

// Issue one async global->LDS 128-bit copy (CDNA5 ASYNCcnt path, no VGPR data).
__device__ __forceinline__ void async_copy_b128(const bf16* gsrc, const bf16* ldst)
{
    unsigned laddr = (unsigned)(size_t)ldst;   // generic LDS ptr low 32 bits == LDS addr
    asm volatile("global_load_async_to_lds_b128 %0, %1, off"
                 :: "v"(laddr), "v"(gsrc) : "memory");
}
__device__ __forceinline__ void wait_async_all()
{
    asm volatile("s_wait_asynccnt 0x0" ::: "memory");
}

// ---------------------------------------------------------------------------
// NT GEMM: C[M x N] = A[M x K] * B[N x K]^T   (A, B bf16 row-major, K inner)
// fp32 accumulation via v_wmma_f32_16x16x32_bf16, double-buffered LDS filled
// by global_load_async_to_lds_b128. Epilogue fused by `mode`:
//   0: outF[m*N+n] = acc + bias[n] (+ resid[m*N+n])
//   1: Q/K head-split: outB[((b*H+h)*L + l)*64 + dh] = bf16(acc + bias[n])
//   2: V transposed:   outB[((b*H+h)*64 + dh)*L + l] = bf16(acc + bias[n])
//   3: scores (batched z=b*H+h): outF[z*L*L+m*L+n] = acc*scale + rel + rel2d + mask[b*L+n]
//   4: ctx merge (batched): outB[(b*L+m)*(H*64) + h*64 + n] = bf16(acc)
//   5: FFN1: outB[m*N+n] = bf16(gelu_exact(acc + bias[n]))
// L is always 1024 in this workload -> token split uses >>10 / &1023.
// ---------------------------------------------------------------------------
template<int BM, int BN>
__global__ __launch_bounds__(256) void gemm_nt_kernel(
    const bf16* __restrict__ Abase, int lda, long long sAz,
    const bf16* __restrict__ Bbase, int ldb, long long sBz,
    const float* __restrict__ bias,
    int M, int N, int K,
    int mode, float scale,
    float* __restrict__ outF, bf16* __restrict__ outB,
    const float* __restrict__ resid,
    const float* __restrict__ rel, const float* __restrict__ rel2d,
    const float* __restrict__ amask, int Hn)
{
    constexpr int WROWS = (BN >= 128) ? 2 : 4;
    constexpr int WCOLS = 8 / WROWS;
    constexpr int WTM = BM / WROWS;   // wave tile M
    constexpr int WTN = BN / WCOLS;   // wave tile N
    constexpr int MT  = WTM / 16;
    constexpr int NT  = WTN / 16;
    constexpr int LSEQ = 1024;

    __shared__ __align__(16) bf16 Al[2][BM * KBP];
    __shared__ __align__(16) bf16 Bl[2][BN * KBP];

    const int z = blockIdx.z;
    const bf16* Ag = Abase + (long long)z * sAz;
    const bf16* Bg = Bbase + (long long)z * sBz;

    const int m0 = blockIdx.y * BM;
    const int n0 = blockIdx.x * BN;

    const int tid  = threadIdx.x;
    const int wid  = tid >> 5;
    const int lane = tid & 31;
    const int lrow = lane & 15;
    const int hlf  = lane >> 4;
    const int wm = wid / WCOLS;
    const int wn = wid % WCOLS;

    // Stage one K-slice of A or B into the given LDS buffer (async, b128 chunks).
    auto stageA = [&](int k0, int buf) {
#pragma unroll
        for (int c = tid; c < BM * 4; c += 256) {
            int row = c >> 2, seg = c & 3;
            async_copy_b128(Ag + (size_t)(m0 + row) * lda + k0 + seg * 8,
                            &Al[buf][row * KBP + seg * 8]);
        }
    };
    auto stageB = [&](int k0, int buf) {
#pragma unroll
        for (int c = tid; c < BN * 4; c += 256) {
            int row = c >> 2, seg = c & 3;
            async_copy_b128(Bg + (size_t)(n0 + row) * ldb + k0 + seg * 8,
                            &Bl[buf][row * KBP + seg * 8]);
        }
    };

    v8f acc[MT][NT];
#pragma unroll
    for (int i = 0; i < MT; ++i)
#pragma unroll
        for (int j = 0; j < NT; ++j)
#pragma unroll
            for (int e = 0; e < 8; ++e) acc[i][j][e] = 0.0f;

    // Prologue: fill buffer 0.
    stageA(0, 0);
    stageB(0, 0);
    wait_async_all();
    __syncthreads();

    int buf = 0;
    for (int k0 = 0; k0 < K; k0 += KB) {
        // Kick off next tile into the other buffer while we compute this one.
        if (k0 + KB < K) {
            stageA(k0 + KB, buf ^ 1);
            stageB(k0 + KB, buf ^ 1);
        }

        // Fragment gathers per ISA 7.12.2 VGPR layouts (16-bit A 16x32 / B 32x16).
        FragBF fa[MT], fb[NT];
#pragma unroll
        for (int i = 0; i < MT; ++i) {
            const bf16* ap = &Al[buf][(wm * WTM + i * 16 + lrow) * KBP];
            fa[i].h[0] = *(const v8bf*)(ap + hlf * 8);        // K group 0
            fa[i].h[1] = *(const v8bf*)(ap + 16 + hlf * 8);   // K group 1
        }
#pragma unroll
        for (int j = 0; j < NT; ++j) {
            const bf16* bp = &Bl[buf][(wn * WTN + j * 16 + lrow) * KBP];
            fb[j].h[0] = *(const v8bf*)(bp + hlf * 8);
            fb[j].h[1] = *(const v8bf*)(bp + 16 + hlf * 8);
        }
#pragma unroll
        for (int i = 0; i < MT; ++i)
#pragma unroll
            for (int j = 0; j < NT; ++j)
                acc[i][j] = __builtin_amdgcn_wmma_f32_16x16x32_bf16(
                    false, fa[i].v, false, fb[j].v, (short)0, acc[i][j],
                    false, false);

        // My async copies must have landed, and everyone must be done reading
        // the buffer they are about to overwrite next iteration.
        wait_async_all();
        __syncthreads();
        buf ^= 1;
    }

    // Epilogue. C layout: lane col = lane&15, VGPR e -> row e + 8*(lane>=16).
#pragma unroll
    for (int i = 0; i < MT; ++i) {
#pragma unroll
        for (int j = 0; j < NT; ++j) {
            const int nn = n0 + wn * WTN + j * 16 + lrow;
#pragma unroll
            for (int e = 0; e < 8; ++e) {
                const int mm = m0 + wm * WTM + i * 16 + e + 8 * hlf;
                if (mm >= M || nn >= N) continue;
                float v = acc[i][j][e];
                switch (mode) {
                    case 0: {
                        v += bias[nn];
                        if (resid) v += resid[(size_t)mm * N + nn];
                        outF[(size_t)mm * N + nn] = v;
                    } break;
                    case 1: {
                        v += bias[nn];
                        int b = mm >> 10, l = mm & (LSEQ - 1), h = nn >> 6, dh = nn & 63;
                        outB[(((size_t)(b * Hn + h)) * LSEQ + l) * 64 + dh] = (bf16)v;
                    } break;
                    case 2: {
                        v += bias[nn];
                        int b = mm >> 10, l = mm & (LSEQ - 1), h = nn >> 6, dh = nn & 63;
                        outB[(((size_t)(b * Hn + h)) * 64 + dh) * LSEQ + l] = (bf16)v;
                    } break;
                    case 3: {
                        size_t idx = ((size_t)z << 20) + ((size_t)mm << 10) + nn;
                        int b = z / Hn;
                        outF[idx] = v * scale + rel[idx] + rel2d[idx]
                                    + amask[(size_t)(b << 10) + nn];
                    } break;
                    case 4: {
                        int b = z / Hn, h = z - b * Hn;
                        outB[((size_t)((b << 10) + mm)) * (Hn * 64) + h * 64 + nn] = (bf16)v;
                    } break;
                    case 5: {
                        v += bias[nn];
                        v = 0.5f * v * (1.0f + erff(v * 0.70710678118654752f));
                        outB[(size_t)mm * N + nn] = (bf16)v;
                    } break;
                    default: break;
                }
            }
        }
    }
}

// Row softmax over len<=1024 fp32 -> bf16 probabilities.
__global__ __launch_bounds__(256) void softmax_kernel(
    const float* __restrict__ S, bf16* __restrict__ P, int len)
{
    __shared__ float red[256];
    const size_t base = (size_t)blockIdx.x * len;
    float vals[4];
    float mx = -3.4e38f;
#pragma unroll
    for (int i = 0; i < 4; ++i) {
        int j = threadIdx.x + i * 256;
        vals[i] = (j < len) ? S[base + j] : -3.4e38f;
        mx = fmaxf(mx, vals[i]);
    }
    red[threadIdx.x] = mx; __syncthreads();
    for (int s = 128; s > 0; s >>= 1) {
        if (threadIdx.x < s) red[threadIdx.x] = fmaxf(red[threadIdx.x], red[threadIdx.x + s]);
        __syncthreads();
    }
    mx = red[0]; __syncthreads();
    float sum = 0.f;
#pragma unroll
    for (int i = 0; i < 4; ++i) { vals[i] = __expf(vals[i] - mx); sum += vals[i]; }
    red[threadIdx.x] = sum; __syncthreads();
    for (int s = 128; s > 0; s >>= 1) {
        if (threadIdx.x < s) red[threadIdx.x] += red[threadIdx.x + s];
        __syncthreads();
    }
    const float inv = 1.0f / red[0];
#pragma unroll
    for (int i = 0; i < 4; ++i) {
        int j = threadIdx.x + i * 256;
        if (j < len) P[base + j] = (bf16)(vals[i] * inv);
    }
}

// Per-row LayerNorm over D elements; writes fp32 (and bf16 copy for next GEMM).
__global__ __launch_bounds__(256) void layernorm_kernel(
    const float* __restrict__ y, const float* __restrict__ gw,
    const float* __restrict__ bw, float* __restrict__ outF,
    bf16* __restrict__ outB, int Dd)
{
    __shared__ float s1[256];
    __shared__ float s2[256];
    const int row = blockIdx.x;
    const float* x = y + (size_t)row * Dd;
    float a = 0.f, b = 0.f;
    for (int j = threadIdx.x; j < Dd; j += 256) { float t = x[j]; a += t; b += t * t; }
    s1[threadIdx.x] = a; s2[threadIdx.x] = b;
    __syncthreads();
    for (int s = 128; s > 0; s >>= 1) {
        if (threadIdx.x < s) {
            s1[threadIdx.x] += s1[threadIdx.x + s];
            s2[threadIdx.x] += s2[threadIdx.x + s];
        }
        __syncthreads();
    }
    const float mean = s1[0] / Dd;
    const float var  = s2[0] / Dd - mean * mean;
    const float rs   = rsqrtf(var + 1e-12f);
    for (int j = threadIdx.x; j < Dd; j += 256) {
        float t = (x[j] - mean) * rs * gw[j] + bw[j];
        outF[(size_t)row * Dd + j] = t;
        if (outB) outB[(size_t)row * Dd + j] = (bf16)t;
    }
}

// fp32 [K][N] -> bf16 [N][K] (transpose + convert; run once per weight).
__global__ void transpose_convert_kernel(const float* __restrict__ in,
                                         bf16* __restrict__ out, int K, int N)
{
    size_t idx = (size_t)blockIdx.x * blockDim.x + threadIdx.x;
    if (idx >= (size_t)K * N) return;
    int k = (int)(idx / N), n = (int)(idx % N);
    out[(size_t)n * K + k] = (bf16)in[idx];
}

__global__ void convert_copy_kernel(const float* __restrict__ in,
                                    float* __restrict__ outF,
                                    bf16* __restrict__ outB, long long n)
{
    long long idx = (long long)blockIdx.x * blockDim.x + threadIdx.x;
    if (idx >= n) return;
    float v = in[idx];
    outF[idx] = v;
    outB[idx] = (bf16)v;
}

extern "C" void kernel_launch(void* const* d_in, const int* in_sizes, int n_in,
                              void* d_out, int out_size, void* d_ws, size_t ws_size,
                              hipStream_t stream)
{
    (void)in_sizes; (void)n_in; (void)out_size; (void)ws_size;
    const int Bv = 2, Lv = 1024, Hv = 12, DHv = 64, Dv = 768, FFv = 3072, NLv = 4;
    const int Mv = Bv * Lv;     // 2048 tokens
    const int BH = Bv * Hv;     // 24 (batch*heads)
    const float SCALE = 0.125f; // 1/sqrt(64)

    const float* hidden = (const float*)d_in[0];
    const float* amask  = (const float*)d_in[1];
    const float* rel    = (const float*)d_in[2];
    const float* rel2d  = (const float*)d_in[3];
    const float* wq = (const float*)d_in[4];
    const float* bq = (const float*)d_in[5];
    const float* wk = (const float*)d_in[6];
    const float* bk = (const float*)d_in[7];
    const float* wv = (const float*)d_in[8];
    const float* bv = (const float*)d_in[9];
    const float* wo = (const float*)d_in[10];
    const float* bo = (const float*)d_in[11];
    const float* ln1g = (const float*)d_in[12];
    const float* ln1b = (const float*)d_in[13];
    const float* w1 = (const float*)d_in[14];
    const float* b1 = (const float*)d_in[15];
    const float* w2 = (const float*)d_in[16];
    const float* b2 = (const float*)d_in[17];
    const float* ln2g = (const float*)d_in[18];
    const float* ln2b = (const float*)d_in[19];

    char* ws = (char*)d_ws;
    size_t off = 0;
    auto alloc = [&](size_t bytes) -> void* {
        void* p = ws + off;
        off = (off + bytes + 255) & ~(size_t)255;
        return p;
    };

    // bf16 transposed weights (persist across layers)
    bf16* wqT = (bf16*)alloc((size_t)NLv * Dv * Dv * sizeof(bf16));
    bf16* wkT = (bf16*)alloc((size_t)NLv * Dv * Dv * sizeof(bf16));
    bf16* wvT = (bf16*)alloc((size_t)NLv * Dv * Dv * sizeof(bf16));
    bf16* woT = (bf16*)alloc((size_t)NLv * Dv * Dv * sizeof(bf16));
    bf16* w1T = (bf16*)alloc((size_t)NLv * Dv * FFv * sizeof(bf16)); // [FF][D]
    bf16* w2T = (bf16*)alloc((size_t)NLv * FFv * Dv * sizeof(bf16)); // [D][FF]
    // activations
    float* xF    = (float*)alloc((size_t)Mv * Dv * sizeof(float));
    bf16*  xB    = (bf16*) alloc((size_t)Mv * Dv * sizeof(bf16));
    bf16*  qB    = (bf16*) alloc((size_t)Mv * Dv * sizeof(bf16));   // [B,H,L,DH]
    bf16*  kB    = (bf16*) alloc((size_t)Mv * Dv * sizeof(bf16));   // [B,H,L,DH]
    bf16*  vtB   = (bf16*) alloc((size_t)Mv * Dv * sizeof(bf16));   // [B,H,DH,L]
    float* S     = (float*)alloc((size_t)BH * Lv * Lv * sizeof(float));
    bf16*  P     = (bf16*) alloc((size_t)BH * Lv * Lv * sizeof(bf16));
    bf16*  ctxB  = (bf16*) alloc((size_t)Mv * Dv * sizeof(bf16));
    float* yF    = (float*)alloc((size_t)Mv * Dv * sizeof(float));
    float* attnF = (float*)alloc((size_t)Mv * Dv * sizeof(float));
    bf16*  attnB = (bf16*) alloc((size_t)Mv * Dv * sizeof(bf16));
    bf16*  h1B   = (bf16*) alloc((size_t)Mv * FFv * sizeof(bf16));

    const int TPB = 256;
    // One-time weight transpose + bf16 convert
    for (int i = 0; i < NLv; ++i) {
        size_t od = (size_t)i * Dv * Dv;
        transpose_convert_kernel<<<(Dv * Dv + TPB - 1) / TPB, TPB, 0, stream>>>(wq + od, wqT + od, Dv, Dv);
        transpose_convert_kernel<<<(Dv * Dv + TPB - 1) / TPB, TPB, 0, stream>>>(wk + od, wkT + od, Dv, Dv);
        transpose_convert_kernel<<<(Dv * Dv + TPB - 1) / TPB, TPB, 0, stream>>>(wv + od, wvT + od, Dv, Dv);
        transpose_convert_kernel<<<(Dv * Dv + TPB - 1) / TPB, TPB, 0, stream>>>(wo + od, woT + od, Dv, Dv);
        size_t of = (size_t)i * Dv * FFv;
        transpose_convert_kernel<<<(Dv * FFv + TPB - 1) / TPB, TPB, 0, stream>>>(w1 + of, w1T + of, Dv, FFv);
        transpose_convert_kernel<<<(FFv * Dv + TPB - 1) / TPB, TPB, 0, stream>>>(w2 + of, w2T + of, FFv, Dv);
    }
    convert_copy_kernel<<<((long long)Mv * Dv + TPB - 1) / TPB, TPB, 0, stream>>>(
        hidden, xF, xB, (long long)Mv * Dv);

    for (int i = 0; i < NLv; ++i) {
        const bf16* wqt = wqT + (size_t)i * Dv * Dv;
        const bf16* wkt = wkT + (size_t)i * Dv * Dv;
        const bf16* wvt = wvT + (size_t)i * Dv * Dv;
        const bf16* wot = woT + (size_t)i * Dv * Dv;
        const bf16* w1t = w1T + (size_t)i * Dv * FFv;
        const bf16* w2t = w2T + (size_t)i * FFv * Dv;

        // Q/K projections -> head-split bf16 [B,H,L,DH]
        gemm_nt_kernel<128,128><<<dim3(Dv/128, Mv/128, 1), 256, 0, stream>>>(
            xB, Dv, 0, wqt, Dv, 0, bq + (size_t)i * Dv, Mv, Dv, Dv,
            1, 1.0f, nullptr, qB, nullptr, nullptr, nullptr, nullptr, Hv);
        gemm_nt_kernel<128,128><<<dim3(Dv/128, Mv/128, 1), 256, 0, stream>>>(
            xB, Dv, 0, wkt, Dv, 0, bk + (size_t)i * Dv, Mv, Dv, Dv,
            1, 1.0f, nullptr, kB, nullptr, nullptr, nullptr, nullptr, Hv);
        // V projection -> transposed bf16 [B,H,DH,L]
        gemm_nt_kernel<128,128><<<dim3(Dv/128, Mv/128, 1), 256, 0, stream>>>(
            xB, Dv, 0, wvt, Dv, 0, bv + (size_t)i * Dv, Mv, Dv, Dv,
            2, 1.0f, nullptr, vtB, nullptr, nullptr, nullptr, nullptr, Hv);
        // scores = Q K^T * scale + rel + rel2d + mask   (batched over B*H)
        gemm_nt_kernel<128,128><<<dim3(Lv/128, Lv/128, BH), 256, 0, stream>>>(
            qB, DHv, (long long)Lv * DHv, kB, DHv, (long long)Lv * DHv, nullptr,
            Lv, Lv, DHv, 3, SCALE, S, nullptr, nullptr, rel, rel2d, amask, Hv);
        softmax_kernel<<<BH * Lv, 256, 0, stream>>>(S, P, Lv);
        // ctx = P * V = P * (V^T)^T  (NT with B = vt [DH][L]), merge heads
        gemm_nt_kernel<128,64><<<dim3(DHv/64, Lv/128, BH), 256, 0, stream>>>(
            P, Lv, (long long)Lv * Lv, vtB, Lv, (long long)DHv * Lv, nullptr,
            Lv, DHv, Lv, 4, 1.0f, nullptr, ctxB, nullptr, nullptr, nullptr, nullptr, Hv);
        // attn out: ctx*Wo + bo + x  -> LN1
        gemm_nt_kernel<128,128><<<dim3(Dv/128, Mv/128, 1), 256, 0, stream>>>(
            ctxB, Dv, 0, wot, Dv, 0, bo + (size_t)i * Dv, Mv, Dv, Dv,
            0, 1.0f, yF, nullptr, xF, nullptr, nullptr, nullptr, Hv);
        layernorm_kernel<<<Mv, 256, 0, stream>>>(
            yF, ln1g + (size_t)i * Dv, ln1b + (size_t)i * Dv, attnF, attnB, Dv);
        // FFN1 + exact GELU
        gemm_nt_kernel<128,128><<<dim3(FFv/128, Mv/128, 1), 256, 0, stream>>>(
            attnB, Dv, 0, w1t, Dv, 0, b1 + (size_t)i * FFv, Mv, FFv, Dv,
            5, 1.0f, nullptr, h1B, nullptr, nullptr, nullptr, nullptr, Hv);
        // FFN2 + residual -> LN2
        gemm_nt_kernel<128,128><<<dim3(Dv/128, Mv/128, 1), 256, 0, stream>>>(
            h1B, FFv, 0, w2t, FFv, 0, b2 + (size_t)i * Dv, Mv, Dv, FFv,
            0, 1.0f, yF, nullptr, attnF, nullptr, nullptr, nullptr, Hv);
        float* lnOut = (i == NLv - 1) ? (float*)d_out : xF;
        layernorm_kernel<<<Mv, 256, 0, stream>>>(
            yF, ln2g + (size_t)i * Dv, ln2b + (size_t)i * Dv, lnOut, xB, Dv);
    }
}
